// ContrastiveLoss_49160195670085
// MI455X (gfx1250) — matrix operand order repaired and checked by reading, not compiled, not run on hardware
//
#include <hip/hip_runtime.h>
#include <hip/hip_bf16.h>

// ---- types matching the gfx1250 WMMA builtin signatures ----
typedef __attribute__((ext_vector_type(16))) __bf16 v16bf;
typedef __attribute__((ext_vector_type(8)))  __bf16 v8bf;
typedef __attribute__((ext_vector_type(4)))  __bf16 v4bf;
typedef __attribute__((ext_vector_type(8)))  float  v8f;

#define NROWS 8192
#define DDIM  128
#define MB 32                    // output rows per wave-job
#define NB 64                    // output cols per wave-job
#define JOBS_M (NROWS / MB)      // 256
#define JOBS_N (NROWS / NB)      // 128
#define NJOBS  (JOBS_M * JOBS_N) // 32768
#define WAVES_PER_BLOCK 8

// --------------------------------------------------------------------------
// Kernel 1: L2-normalize each row, split into bf16 hi/lo pair (hi+lo ~= fp32)
// One wave (32 lanes) per row; each lane handles 4 contiguous floats.
// --------------------------------------------------------------------------
__global__ void __launch_bounds__(256)
norm_split_kernel(const float* __restrict__ emb,
                  __bf16* __restrict__ Hi, __bf16* __restrict__ Lo)
{
    const int wave = threadIdx.x >> 5;
    const int lane = threadIdx.x & 31;
    const int row  = blockIdx.x * WAVES_PER_BLOCK + wave;

    const float4 v = *(const float4*)(emb + (size_t)row * DDIM + lane * 4);
    float s = v.x * v.x + v.y * v.y + v.z * v.z + v.w * v.w;
    #pragma unroll
    for (int off = 16; off > 0; off >>= 1)
        s += __shfl_xor(s, off, 32);

    const float rinv = 1.0f / fmaxf(sqrtf(s), 1e-12f);
    float e[4] = { v.x * rinv, v.y * rinv, v.z * rinv, v.w * rinv };

    v4bf hi, lo;
    #pragma unroll
    for (int i = 0; i < 4; ++i) {
        __bf16 h = (__bf16)e[i];
        hi[i] = h;
        lo[i] = (__bf16)(e[i] - (float)h);
    }
    const size_t o = (size_t)row * DDIM + lane * 4;
    *(v4bf*)(Hi + o) = hi;
    *(v4bf*)(Lo + o) = lo;
}

// --------------------------------------------------------------------------
// Fragment loaders matching CDNA5 16-bit WMMA VGPR layouts (ISA 7.12.2).
// A (16x32, MxK): lanes 0-15 -> M=lane, VGPR0-3 = K 0..7, VGPR4-7 = K 16..23
//                 lanes 16-31 -> same M, K 8..15 and 24..31.
// B (32x16, KxN): lanes 0-15 -> N=lane, VGPR0-3 = K 0..7, VGPR4-7 = K 16..23
//                 lanes 16-31 -> same N, K 8..15 and 24..31.
// --------------------------------------------------------------------------
__device__ __forceinline__ v16bf load_a_frag(const __bf16* __restrict__ base,
                                             int row, int kbase, int hb)
{
    const __bf16* p = base + (size_t)row * DDIM + kbase + hb * 8;
    v8bf c0 = *(const v8bf*)(p);        // K offsets 0..7   (or 8..15)
    v8bf c1 = *(const v8bf*)(p + 16);   // K offsets 16..23 (or 24..31)
    v16bf f;
    #pragma unroll
    for (int i = 0; i < 8; ++i) { f[i] = c0[i]; f[8 + i] = c1[i]; }
    return f;
}

__device__ __forceinline__ v16bf load_b_frag(const __bf16* __restrict__ base,
                                             int row, int kbase, int hb)
{
    const __bf16* p = base + (size_t)row * DDIM + kbase + hb * 8;
    v8bf c0 = *(const v8bf*)(p);
    v8bf c1 = *(const v8bf*)(p + 16);
    v16bf f;
    #pragma unroll
    for (int i = 0; i < 8; ++i) { f[i] = c0[i]; f[8 + i] = c1[i]; }
    return f;
}

__device__ __forceinline__ v8f wmma_bf16(v16bf a, v16bf b, v8f c)
{
    return __builtin_amdgcn_wmma_f32_16x16x32_bf16(
        /*neg_a=*/false, a, /*neg_b=*/false, b,
        /*c_mod=*/(short)0, c, /*reuse_a=*/false, /*reuse_b=*/false);
}

// --------------------------------------------------------------------------
// Kernel 2: fused GEMM (split-bf16 ~ fp32) + contrastive loss epilogue.
// Each wave computes a 32x64 block of sim = E*E^T (2x4 tiles of 16x16),
// reads the matching labels, accumulates sum(loss) and count(loss>0),
// writes one deterministic partial per wave-job.
// --------------------------------------------------------------------------
__global__ void __launch_bounds__(256)
gemm_loss_kernel(const __bf16* __restrict__ Hi, const __bf16* __restrict__ Lo,
                 const int* __restrict__ label,
                 float* __restrict__ pSum, unsigned* __restrict__ pCnt)
{
    const int wave = threadIdx.x >> 5;
    const int lane = threadIdx.x & 31;
    const int hb   = lane >> 4;      // half-wave id
    const int l16  = lane & 15;

    const int job   = blockIdx.x * WAVES_PER_BLOCK + wave;
    const int tm    = job / JOBS_N;
    const int tn    = job % JOBS_N;
    const int mbase = tm * MB;
    const int nbase = tn * NB;

    // prefetch the label block rows (32 rows, one per lane)
    __builtin_prefetch(&label[(size_t)(mbase + lane) * NROWS + nbase], 0, 1);

    v8f acc[2][4];
    #pragma unroll
    for (int i = 0; i < 2; ++i)
        #pragma unroll
        for (int j = 0; j < 4; ++j)
            acc[i][j] = (v8f)(0.0f);

    #pragma unroll
    for (int ks = 0; ks < DDIM / 32; ++ks) {
        const int kbase = ks * 32;
        v16bf ah[2], al[2], bh[4], bl[4];
        #pragma unroll
        for (int i = 0; i < 2; ++i) {
            ah[i] = load_a_frag(Hi, mbase + i * 16 + l16, kbase, hb);
            al[i] = load_a_frag(Lo, mbase + i * 16 + l16, kbase, hb);
        }
        #pragma unroll
        for (int j = 0; j < 4; ++j) {
            bh[j] = load_b_frag(Hi, nbase + j * 16 + l16, kbase, hb);
            bl[j] = load_b_frag(Lo, nbase + j * 16 + l16, kbase, hb);
        }
        #pragma unroll
        for (int i = 0; i < 2; ++i)
            #pragma unroll
            for (int j = 0; j < 4; ++j) {
                acc[i][j] = wmma_bf16(ah[i], bh[j], acc[i][j]);  // hi*hi
                acc[i][j] = wmma_bf16(al[i], bh[j], acc[i][j]);  // lo*hi
                acc[i][j] = wmma_bf16(ah[i], bl[j], acc[i][j]);  // hi*lo
            }
    }

    // Epilogue: C-matrix layout (32-bit 16x16): lanes 0-15 N=l16 M=r,
    // lanes 16-31 N=l16 M=r+8, VGPR r = 0..7.
    float lsum = 0.0f;
    int   lcnt = 0;
    #pragma unroll
    for (int i = 0; i < 2; ++i) {
        #pragma unroll
        for (int j = 0; j < 4; ++j) {
            #pragma unroll
            for (int r = 0; r < 8; ++r) {
                const int m = mbase + i * 16 + (hb ? r + 8 : r);
                const int n = nbase + j * 16 + l16;
                const int lab = label[(size_t)m * NROWS + n];
                const float sim = acc[i][j][r];
                // dis = 1 - sim; pos = relu(dis - 0); neg = relu(1 - dis) = relu(sim)
                const float pos = fmaxf(1.0f - sim, 0.0f);
                const float neg = fmaxf(sim, 0.0f);
                const float lv  = lab ? 0.5f * pos * pos : 0.5f * neg * neg;
                lsum += lv;
                lcnt += (lv > 0.0f) ? 1 : 0;
            }
        }
    }

    // wave32 shuffle reduction (fixed order -> deterministic)
    #pragma unroll
    for (int off = 16; off > 0; off >>= 1) {
        lsum += __shfl_xor(lsum, off, 32);
        lcnt += __shfl_xor(lcnt, off, 32);
    }
    if (lane == 0) {
        pSum[job] = lsum;
        pCnt[job] = (unsigned)lcnt;
    }
}

// --------------------------------------------------------------------------
// Kernel 3: deterministic tree reduction of 32768 partials -> scalar loss.
// --------------------------------------------------------------------------
__global__ void __launch_bounds__(1024)
finalize_kernel(const float* __restrict__ pSum, const unsigned* __restrict__ pCnt,
                float* __restrict__ out)
{
    __shared__ double             sL[1024];
    __shared__ unsigned long long cL[1024];
    const int tid = threadIdx.x;

    double s = 0.0;
    unsigned long long c = 0;
    #pragma unroll
    for (int i = 0; i < NJOBS / 1024; ++i) {
        s += (double)pSum[i * 1024 + tid];
        c += (unsigned long long)pCnt[i * 1024 + tid];
    }
    sL[tid] = s; cL[tid] = c;
    __syncthreads();
    for (int st = 512; st > 0; st >>= 1) {
        if (tid < st) { sL[tid] += sL[tid + st]; cL[tid] += cL[tid + st]; }
        __syncthreads();
    }
    if (tid == 0) {
        const double sum = sL[0];
        const unsigned long long nz = cL[0];
        const double loss = (nz > 0)
            ? sum / (double)nz
            : sum / ((double)NROWS * (double)NROWS);  // mean fallback
        out[0] = (float)loss;
    }
}

// --------------------------------------------------------------------------
extern "C" void kernel_launch(void* const* d_in, const int* in_sizes, int n_in,
                              void* d_out, int out_size, void* d_ws, size_t ws_size,
                              hipStream_t stream)
{
    const float* emb   = (const float*)d_in[0];
    const int*   label = (const int*)d_in[1];
    float*       out   = (float*)d_out;

    char* ws = (char*)d_ws;
    __bf16*   Hi   = (__bf16*)(ws);
    __bf16*   Lo   = (__bf16*)(ws + (size_t)NROWS * DDIM * sizeof(__bf16));
    float*    pSum = (float*)(ws + (size_t)NROWS * DDIM * 2 * sizeof(__bf16));
    unsigned* pCnt = (unsigned*)(ws + (size_t)NROWS * DDIM * 2 * sizeof(__bf16)
                                 + (size_t)NJOBS * sizeof(float));

    norm_split_kernel<<<NROWS / WAVES_PER_BLOCK, 256, 0, stream>>>(emb, Hi, Lo);
    gemm_loss_kernel<<<NJOBS / WAVES_PER_BLOCK, 256, 0, stream>>>(Hi, Lo, label, pSum, pCnt);
    finalize_kernel<<<1, 1024, 0, stream>>>(pSum, pCnt, out);
}
